// fusion_model_graph_34608846471590
// MI455X (gfx1250) — matrix operand order repaired and checked by reading, not compiled, not run on hardware
//
#include <hip/hip_runtime.h>
#include <stdint.h>

typedef __attribute__((ext_vector_type(16))) __bf16 bf16x16;
typedef __attribute__((ext_vector_type(8)))  float  f32x8;

static __device__ __forceinline__ int imin(int a, int b) { return a < b ? a : b; }

// ---------------------------------------------------------------------------
// Zero scratch accumulators
// ---------------------------------------------------------------------------
__global__ __launch_bounds__(256) void zero_kernel(float* __restrict__ p, size_t n) {
    size_t i = (size_t)blockIdx.x * 256 + threadIdx.x;
    if (i < n) p[i] = 0.0f;
}

// ---------------------------------------------------------------------------
// Pack a row-major f32 [K x N] matrix into WMMA-B fragment-major bf16 layout.
// B order used here (contiguous k, as A):  lane l: n = l&15, k = (l>>4)*16 + i,
// i = 0..15.  Packed index = ((kt*ntiles + nt)*32 + lane)*16 + i  -> each lane
// reads one contiguous, 32B-aligned 32-byte chunk in the GEMM.
// ---------------------------------------------------------------------------
__global__ __launch_bounds__(256) void packB_kernel(const float* __restrict__ W,
                                                    __bf16* __restrict__ out,
                                                    int Kdim, int Ncols) {
    size_t idx = (size_t)blockIdx.x * 256 + threadIdx.x;
    size_t tot = (size_t)Kdim * Ncols;
    if (idx >= tot) return;
    int   i    = (int)(idx & 15);
    int   lane = (int)((idx >> 4) & 31);
    size_t t   = idx >> 9;
    int ntiles = Ncols >> 4;
    int nt = (int)(t % ntiles);
    int kt = (int)(t / ntiles);
    int k = (kt << 5) + ((lane >> 4) << 4) + i;
    int n = (nt << 4) + (lane & 15);
    out[idx] = (__bf16)W[(size_t)k * Ncols + n];
}

// ---------------------------------------------------------------------------
// WMMA GEMM:  C[M x N] = A[M x K](f32, converted to bf16 on the fly) * Bpack
//             (+ bias).  Block = 256 thr = 8 waves; block tile 128x128;
//             wave tile 32x128? no: wave = 2(M)x4(N) 16x16 tiles = 32x64.
// A-frag per lane: rows m=lane&15 (+16 for mi=1), K runs
//   [k0, k0+16) with k0 = kt*32 + (lane>>4)*16.
// ---------------------------------------------------------------------------
union AFrag { bf16x16 v; __bf16 e[16]; };

__global__ __launch_bounds__(256)
void gemm_bf16_wmma(const float* __restrict__ A, const __bf16* __restrict__ Bpack,
                    const float* __restrict__ bias, float* __restrict__ C,
                    int M, int Kdim, int Ncols) {
    const int tid   = threadIdx.x;
    const int lane  = tid & 31;
    const int wave  = tid >> 5;
    const int waveM = wave & 3;   // 0..3
    const int waveN = wave >> 2;  // 0..1
    const int m16   = lane & 15;
    const int half  = lane >> 4;

    const int rowBase = blockIdx.y * 128 + waveM * 32;
    const int colBase = blockIdx.x * 128 + waveN * 64;
    const int ntiles  = Ncols >> 4;
    const int ktiles  = Kdim >> 5;
    const int ntBase  = colBase >> 4;

    f32x8 acc[2][4] = {};

    int arow0 = imin(rowBase + m16, M - 1);
    int arow1 = imin(rowBase + 16 + m16, M - 1);
    const float* aptr0 = A + (size_t)arow0 * Kdim;
    const float* aptr1 = A + (size_t)arow1 * Kdim;

    for (int kt = 0; kt < ktiles; ++kt) {
        const int k0 = (kt << 5) + (half << 4);
        AFrag af[2];
#pragma unroll
        for (int mi = 0; mi < 2; ++mi) {
            const float* ap = (mi ? aptr1 : aptr0) + k0;
            float4 x0 = *(const float4*)(ap);
            float4 x1 = *(const float4*)(ap + 4);
            float4 x2 = *(const float4*)(ap + 8);
            float4 x3 = *(const float4*)(ap + 12);
            af[mi].e[0]  = (__bf16)x0.x; af[mi].e[1]  = (__bf16)x0.y;
            af[mi].e[2]  = (__bf16)x0.z; af[mi].e[3]  = (__bf16)x0.w;
            af[mi].e[4]  = (__bf16)x1.x; af[mi].e[5]  = (__bf16)x1.y;
            af[mi].e[6]  = (__bf16)x1.z; af[mi].e[7]  = (__bf16)x1.w;
            af[mi].e[8]  = (__bf16)x2.x; af[mi].e[9]  = (__bf16)x2.y;
            af[mi].e[10] = (__bf16)x2.z; af[mi].e[11] = (__bf16)x2.w;
            af[mi].e[12] = (__bf16)x3.x; af[mi].e[13] = (__bf16)x3.y;
            af[mi].e[14] = (__bf16)x3.z; af[mi].e[15] = (__bf16)x3.w;
        }
#pragma unroll
        for (int ni = 0; ni < 4; ++ni) {
            const __bf16* bp =
                Bpack + ((size_t)(kt * ntiles + ntBase + ni) * 32 + lane) * 16;
            bf16x16 bf = *(const bf16x16*)bp;
            // prefetch same fragment for next K-step (resident-friendly: B is
            // small and hot in the 192MB L2; this pulls it toward the WGP).
            __builtin_prefetch(bp + (size_t)ntiles * 512, 0, 3);
            acc[0][ni] = __builtin_amdgcn_wmma_f32_16x16x32_bf16(
                false, af[0].v, false, bf, (short)0, acc[0][ni], false, false);
            acc[1][ni] = __builtin_amdgcn_wmma_f32_16x16x32_bf16(
                false, af[1].v, false, bf, (short)0, acc[1][ni], false, false);
        }
    }

    // C layout: vgpr r -> row = tileRow + (lane>>4)*8 + r, col = tileCol + (lane&15)
#pragma unroll
    for (int mi = 0; mi < 2; ++mi) {
        const int r0 = rowBase + mi * 16 + half * 8;
#pragma unroll
        for (int ni = 0; ni < 4; ++ni) {
            const int col = colBase + ni * 16 + m16;
            const float b = bias ? bias[col] : 0.0f;
#pragma unroll
            for (int r = 0; r < 8; ++r) {
                int row = r0 + r;
                if (row < M) C[(size_t)row * Ncols + col] = acc[mi][ni][r] + b;
            }
        }
    }
}

// ---------------------------------------------------------------------------
// Stats: sumsq(Q), sumsq(K) (global), ks_sum[col], vs_sum[col]
// grid: (N/256, ceil(M/100)); each thread owns one column over 100 rows.
// ---------------------------------------------------------------------------
__global__ __launch_bounds__(256)
void stats_kernel(const float* __restrict__ Q, const float* __restrict__ Kmat,
                  const float* __restrict__ V, float* __restrict__ sumsq,
                  float* __restrict__ ks_sum, float* __restrict__ vs_sum,
                  int M, int D) {
    __shared__ float red[256];
    const int tid = threadIdx.x;
    const int col = blockIdx.x * 256 + tid;
    const int r0  = blockIdx.y * 100;
    float aks = 0.f, avs = 0.f, q2 = 0.f, k2 = 0.f;
    for (int j = 0; j < 100; ++j) {
        int rr = r0 + j;
        if (rr >= M) break;
        size_t idx = (size_t)rr * D + col;
        float q = Q[idx], k = Kmat[idx], v = V[idx];
        aks += k; avs += v; q2 += q * q; k2 += k * k;
    }
    atomicAdd(&ks_sum[col], aks);
    atomicAdd(&vs_sum[col], avs);
    red[tid] = q2; __syncthreads();
    for (int s = 128; s > 0; s >>= 1) {
        if (tid < s) red[tid] += red[tid + s];
        __syncthreads();
    }
    if (tid == 0) atomicAdd(&sumsq[0], red[0]);
    __syncthreads();
    red[tid] = k2; __syncthreads();
    for (int s = 128; s > 0; s >>= 1) {
        if (tid < s) red[tid] += red[tid + s];
        __syncthreads();
    }
    if (tid == 0) atomicAdd(&sumsq[1], red[0]);
}

// ---------------------------------------------------------------------------
// kvs[m,d] = sum_n K[n,m]*V[n,d].  Tiled outer product, split-K atomics.
// Block: 64(m)x64(d) tile, 256 threads, each 4x4 outputs; grid.z n-chunks.
// ---------------------------------------------------------------------------
__global__ __launch_bounds__(256)
void kvs_kernel(const float* __restrict__ Kmat, const float* __restrict__ V,
                float* __restrict__ kvs, int Nrows, int D) {
    __shared__ float sk[8][64];
    __shared__ float sv[8][64];
    const int tid = threadIdx.x;
    const int tx = tid & 15, ty = tid >> 4;
    const int m0 = blockIdx.y * 64, d0 = blockIdx.x * 64;
    const int chunk = (Nrows + gridDim.z - 1) / gridDim.z;
    const int n0 = blockIdx.z * chunk;
    const int n1 = imin(n0 + chunk, Nrows);
    float acc[4][4] = {};
    for (int n = n0; n < n1; n += 8) {
        for (int l = tid; l < 512; l += 256) {
            int j = l >> 6, c = l & 63;
            int rr = n + j;
            float kk = 0.f, vv = 0.f;
            if (rr < n1) {
                kk = Kmat[(size_t)rr * D + m0 + c];
                vv = V[(size_t)rr * D + d0 + c];
            }
            sk[j][c] = kk; sv[j][c] = vv;
        }
        __syncthreads();
#pragma unroll
        for (int j = 0; j < 8; ++j) {
            float kr[4], vr[4];
#pragma unroll
            for (int a = 0; a < 4; ++a) { kr[a] = sk[j][ty * 4 + a]; vr[a] = sv[j][tx * 4 + a]; }
#pragma unroll
            for (int a = 0; a < 4; ++a)
#pragma unroll
                for (int b = 0; b < 4; ++b) acc[a][b] += kr[a] * vr[b];
        }
        __syncthreads();
    }
#pragma unroll
    for (int a = 0; a < 4; ++a)
#pragma unroll
        for (int b = 0; b < 4; ++b)
            atomicAdd(&kvs[(size_t)(m0 + ty * 4 + a) * D + d0 + tx * 4 + b], acc[a][b]);
}

// ---------------------------------------------------------------------------
// r[i] = Q[i,:] . ks_sum  (raw, unnormalized)
// ---------------------------------------------------------------------------
__global__ __launch_bounds__(256)
void rdot_kernel(const float* __restrict__ Q, const float* __restrict__ ks_sum,
                 float* __restrict__ r, int M, int D) {
    __shared__ float ks[1024];
    for (int i = threadIdx.x; i < D; i += 256) ks[i] = ks_sum[i];
    __syncthreads();
    int row = blockIdx.x * 256 + threadIdx.x;
    if (row >= M) return;
    const float* q = Q + (size_t)row * D;
    float acc = 0.f;
    for (int i = 0; i < D; i += 4) {
        float4 x = *(const float4*)(q + i);
        acc += x.x * ks[i] + x.y * ks[i + 1] + x.z * ks[i + 2] + x.w * ks[i + 3];
    }
    r[row] = acc;
}

// ---------------------------------------------------------------------------
// Epilogue: out = (s*T + vs_sum[col]) / (s*r[row] + M),  s = 1/(||Q|| ||K||)
// (norms factored out of the normalized-attention formula)
// ---------------------------------------------------------------------------
__global__ __launch_bounds__(256)
void attn_epilogue(float* __restrict__ out, const float* __restrict__ r,
                   const float* __restrict__ vs_sum, const float* __restrict__ sumsq,
                   int M, int D, float nfloat) {
    size_t idx = (size_t)blockIdx.x * 256 + threadIdx.x;
    if (idx >= (size_t)M * D) return;
    int row = (int)(idx / D);
    int col = (int)(idx % D);
    float s = rsqrtf(sumsq[0]) * rsqrtf(sumsq[1]);
    out[idx] = (s * out[idx] + vs_sum[col]) / (s * r[row] + nfloat);
}

// ---------------------------------------------------------------------------
// GCN pieces
// ---------------------------------------------------------------------------
__global__ __launch_bounds__(256)
void deg_kernel(const long long* __restrict__ ei, float* __restrict__ deg, int E) {
    int e = blockIdx.x * 256 + threadIdx.x;
    if (e < E) atomicAdd(&deg[(int)ei[(size_t)E + e]], 1.0f);
}

__global__ __launch_bounds__(256)
void val_kernel(const long long* __restrict__ ei, const float* __restrict__ ew,
                const float* __restrict__ deg, float* __restrict__ val, int E) {
    int e = blockIdx.x * 256 + threadIdx.x;
    if (e >= E) return;
    int row = (int)ei[e];
    int col = (int)ei[(size_t)E + e];
    float v = ew[e] * sqrtf(1.0f / deg[col]) * sqrtf(1.0f / deg[row]);
    if (!__builtin_isfinite(v)) v = 0.0f;
    val[e] = v;
}

__global__ __launch_bounds__(256)
void scatter_kernel(const long long* __restrict__ ei, const float* __restrict__ val,
                    const float* __restrict__ V, float* __restrict__ out,
                    int E, int D) {
    int e = blockIdx.x;
    float v = val[e];
    if (v == 0.0f) return;
    int row = (int)ei[e];
    int col = (int)ei[(size_t)E + e];
    int d = threadIdx.x * 4;
    float4 x = *(const float4*)(V + (size_t)row * D + d);
    float* o = out + (size_t)col * D + d;
    atomicAdd(o + 0, v * x.x);
    atomicAdd(o + 1, v * x.y);
    atomicAdd(o + 2, v * x.z);
    atomicAdd(o + 3, v * x.w);
}

// ---------------------------------------------------------------------------
extern "C" void kernel_launch(void* const* d_in, const int* in_sizes, int n_in,
                              void* d_out, int out_size, void* d_ws, size_t ws_size,
                              hipStream_t stream) {
    const float* x_q = (const float*)d_in[0];
    const float* x_s = (const float*)d_in[1];
    const float* Wq  = (const float*)d_in[2];
    const float* bq  = (const float*)d_in[3];
    const float* Wk  = (const float*)d_in[4];
    const float* bk  = (const float*)d_in[5];
    const float* Wv  = (const float*)d_in[6];
    const float* bv  = (const float*)d_in[7];
    const float* ew  = (const float*)d_in[8];
    const long long* ei = (const long long*)d_in[9];

    const int Dd = in_sizes[3];          // 1024 (H*D, H==1)
    const int IN = in_sizes[2] / Dd;     // 3072
    const int M  = in_sizes[0] / IN;     // 20000
    const int E  = in_sizes[8];          // 640000

    // ---- workspace layout (floats), all offsets multiple of 64 floats ----
    float* ws     = (float*)d_ws;
    float* sumsq  = ws;                                  // 64 (2 used)
    float* ks_sum = ws + 64;                             // Dd
    float* vs_sum = ks_sum + Dd;                         // Dd
    float* deg    = vs_sum + Dd;                         // degPad
    size_t degPad = (size_t)((M + 63) & ~63);
    float* kvs    = deg + degPad;                        // Dd*Dd
    size_t zeroCount = 64 + 2 * (size_t)Dd + degPad + (size_t)Dd * Dd;
    float* r    = kvs + (size_t)Dd * Dd;                 // degPad
    float* val  = r + degPad;                            // E (padded)
    size_t ePad = (size_t)((E + 63) & ~63);
    float* Q    = val + ePad;                            // M*Dd
    float* Km   = Q + (size_t)M * Dd;
    float* V    = Km + (size_t)M * Dd;
    __bf16* WqP  = (__bf16*)(V + (size_t)M * Dd);
    __bf16* WkP  = WqP + (size_t)IN * Dd;
    __bf16* WvP  = WkP + (size_t)IN * Dd;
    __bf16* kvsP = WvP + (size_t)IN * Dd;

    dim3 blk(256);

    // 1) zero accumulators (sumsq, ks_sum, vs_sum, deg, kvs are contiguous)
    zero_kernel<<<(unsigned)((zeroCount + 255) / 256), blk, 0, stream>>>(ws, zeroCount);

    // 2) pack projection weights into WMMA-B fragment layout (bf16)
    {
        size_t tot = (size_t)IN * Dd;
        unsigned g = (unsigned)((tot + 255) / 256);
        packB_kernel<<<g, blk, 0, stream>>>(Wq, WqP, IN, Dd);
        packB_kernel<<<g, blk, 0, stream>>>(Wk, WkP, IN, Dd);
        packB_kernel<<<g, blk, 0, stream>>>(Wv, WvP, IN, Dd);
    }

    // 3) degree histogram (independent of GEMMs)
    deg_kernel<<<(unsigned)((E + 255) / 256), blk, 0, stream>>>(ei, deg, E);

    // 4) projections: Q/K/V = x @ W + b   (WMMA bf16, f32 accum)
    dim3 g1((unsigned)(Dd / 128), (unsigned)((M + 127) / 128));
    gemm_bf16_wmma<<<g1, blk, 0, stream>>>(x_q, WqP, bq, Q,  M, IN, Dd);
    gemm_bf16_wmma<<<g1, blk, 0, stream>>>(x_s, WkP, bk, Km, M, IN, Dd);
    gemm_bf16_wmma<<<g1, blk, 0, stream>>>(x_s, WvP, bv, V,  M, IN, Dd);

    // 5) global sums / norms
    dim3 gs((unsigned)(Dd / 256), (unsigned)((M + 99) / 100));
    stats_kernel<<<gs, blk, 0, stream>>>(Q, Km, V, sumsq, ks_sum, vs_sum, M, Dd);

    // 6) kvs = K^T V (raw; norm factored out) — split-K tiled FMA
    dim3 gk((unsigned)(Dd / 64), (unsigned)(Dd / 64), 10);
    kvs_kernel<<<gk, blk, 0, stream>>>(Km, V, kvs, M, Dd);

    // 7) pack kvs as WMMA-B, r = Q . ks_sum
    packB_kernel<<<(unsigned)(((size_t)Dd * Dd + 255) / 256), blk, 0, stream>>>(kvs, kvsP, Dd, Dd);
    rdot_kernel<<<(unsigned)((M + 255) / 256), blk, 0, stream>>>(Q, ks_sum, r, M, Dd);

    // 8) T = Q @ kvs  directly into d_out (WMMA), then normalize in place
    gemm_bf16_wmma<<<g1, blk, 0, stream>>>(Q, kvsP, nullptr, (float*)d_out, M, Dd, Dd);
    attn_epilogue<<<(unsigned)(((size_t)M * Dd + 255) / 256), blk, 0, stream>>>(
        (float*)d_out, r, vs_sum, sumsq, M, Dd, (float)M);

    // 9) GCN: edge values then scatter-add into d_out
    val_kernel<<<(unsigned)((E + 255) / 256), blk, 0, stream>>>(ei, ew, deg, val, E);
    scatter_kernel<<<(unsigned)E, blk, 0, stream>>>(ei, val, V, (float*)d_out, E, Dd);
}
